// TestAllReduceRMSNormModel_9947144257741
// MI455X (gfx1250) — matrix operand order repaired and checked by reading, not compiled, not run on hardware
//
#include <hip/hip_runtime.h>
#include <stdint.h>

typedef __bf16 bf16;
typedef __attribute__((ext_vector_type(16))) __bf16 v16bf;
typedef __attribute__((ext_vector_type(8)))  float  v8f;
typedef __attribute__((ext_vector_type(4)))  unsigned int u32x4;
typedef __attribute__((ext_vector_type(8)))  int          i32x8;
typedef __attribute__((ext_vector_type(4)))  int          i32x4;

#define H      1024
#define TROWS  32          // rows per block (2 row-groups of 16 per wave)
#define KSTEP  32
#define EPS    1e-6f

#if defined(__AMDGCN__) && __has_builtin(__builtin_amdgcn_tensor_load_to_lds) && \
    __has_builtin(__builtin_amdgcn_s_wait_tensorcnt)
#define HAVE_TDM 1
#else
#define HAVE_TDM 0
#endif

union AB { v16bf v; uint4 u[2]; };

// ---------------------------------------------------------------------------
// Weight prep: wt[n*H + k] = (bf16) w[k*H + n]   (B^T so K is contiguous)
// ---------------------------------------------------------------------------
__global__ __launch_bounds__(256) void wt_kernel(const float* __restrict__ w,
                                                 bf16* __restrict__ wt) {
    int idx = blockIdx.x * 256 + threadIdx.x;   // idx = n*H + k
    int n = idx >> 10;
    int k = idx & (H - 1);
    wt[idx] = (bf16)w[k * H + n];
}

// ---------------------------------------------------------------------------
// Kernel 1: z = relu(x); resid = z; y = rmsnorm(z, g0)  (one block per row)
// ---------------------------------------------------------------------------
__global__ __launch_bounds__(256) void relu_rms_kernel(const float* __restrict__ x,
                                                       const float* __restrict__ g,
                                                       float* __restrict__ resid,
                                                       bf16* __restrict__ y) {
    const int row = blockIdx.x;
    const int tid = threadIdx.x;
    const float* xr = x + (size_t)row * H;

    float v[4];
    float ss = 0.f;
#pragma unroll
    for (int i = 0; i < 4; ++i) {
        float t = xr[tid + 256 * i];
        t = t > 0.f ? t : 0.f;
        v[i] = t;
        ss += t * t;
    }

    __shared__ float red[256];
    red[tid] = ss;
    __syncthreads();
#pragma unroll
    for (int s = 128; s > 0; s >>= 1) {
        if (tid < s) red[tid] += red[tid + s];
        __syncthreads();
    }
    const float scale = rsqrtf(red[0] * (1.0f / H) + EPS);

    float* rr = resid + (size_t)row * H;
    bf16*  yr = y     + (size_t)row * H;
#pragma unroll
    for (int i = 0; i < 4; ++i) {
        int c = tid + 256 * i;
        rr[c] = v[i];
        yr[c] = (bf16)(v[i] * scale * g[c]);
    }
}

// ---------------------------------------------------------------------------
// TDM: DMA one B^T tile (1024 rows x 32 K bf16 = 64KB) into LDS.
// D# per CDNA5 ISA ch.8: 2D tensor, dim0=K (len 1024, stride 1024), tile 32x1024,
// data_size=2B, rows packed contiguously (32 elems/row) at lds_addr.
// ---------------------------------------------------------------------------
#if HAVE_TDM
__device__ __forceinline__ void tdm_load_btile(const bf16* gsrc, bf16* ldsDst) {
    unsigned long long ga = (unsigned long long)(uintptr_t)gsrc;
    unsigned ldsOff = (unsigned)(uintptr_t)ldsDst;   // flat LDS addr[31:0] == LDS offset
    u32x4 g0;
    g0[0] = 1u;                                            // count=1, no gather
    g0[1] = ldsOff;                                        // lds_addr
    g0[2] = (unsigned)(ga & 0xffffffffu);                  // global_addr[31:0]
    g0[3] = (unsigned)((ga >> 32) & 0x01ffffffu)           // global_addr[56:32]
          | (2u << 30);                                    // type = 2 ("image")
    i32x8 g1;
    g1[0] = (int)(1u << 16);          // workgroup_mask=0, data_size=1 (2 bytes)
    g1[1] = (int)(1024u << 16);       // atomic_barrier_addr=0 ; tensor_dim0[15:0]=1024
    g1[2] = (int)(1024u << 16);       // tensor_dim0[31:16]=0  ; tensor_dim1[15:0]=1024
    g1[3] = (int)(32u << 16);         // tensor_dim1[31:16]=0  ; tile_dim0=32
    g1[4] = 1024;                     // tile_dim1=1024 ; tile_dim2=0
    g1[5] = 1024;                     // tensor_dim0_stride[31:0]=1024
    g1[6] = 0;                        // stride hi, tensor_dim1_stride lo
    g1[7] = 0;
    i32x4 gz4 = {0, 0, 0, 0};
    i32x8 gz8 = {0, 0, 0, 0, 0, 0, 0, 0};
    __builtin_amdgcn_tensor_load_to_lds(g0, g1, gz4, gz4, gz8, 0);
}
#endif

// ---------------------------------------------------------------------------
// Load both row-group A fragments for K-chunk k0 (ISA 16-bit A layout, wave32):
// lane L: row = L%16, K = (L/16)*8 + {0..7} and {16..23}.
// ---------------------------------------------------------------------------
__device__ __forceinline__ void load_afrag(AB* a, const bf16* __restrict__ y,
                                           int row0, int lane, int k0) {
#pragma unroll
    for (int gi = 0; gi < 2; ++gi) {
        const bf16* abase =
            y + (size_t)(row0 + 16 * gi + (lane & 15)) * H + k0 + (lane >> 4) * 8;
        a[gi].u[0] = *(const uint4*)(abase);
        a[gi].u[1] = *(const uint4*)(abase + 16);
    }
}

// ---------------------------------------------------------------------------
// One K-step: prefetch next B tile via TDM, prefetch next A fragments into
// registers, then 16 WMMAs per wave (8 column tiles x 2 row groups) with
// explicit bfrag ping-pong so ds_load latency is covered by the previous
// WMMA pair. HN = 1 while another K-step follows.
// ---------------------------------------------------------------------------
template <int HN>
__device__ __forceinline__ void gemm_step(const bf16* __restrict__ y,
                                          const bf16* __restrict__ wt,
                                          bf16* ldsCur, bf16* ldsNext,
                                          int k0, int row0, int lane, int wave,
                                          AB* aCur, AB* aNext,
                                          v8f (*acc)[8]) {
#if HAVE_TDM
    if (wave == 0) {
        if (HN) tdm_load_btile(wt + (size_t)(k0 + KSTEP), ldsNext);
        __builtin_amdgcn_s_wait_tensorcnt(HN ? 1 : 0);
    }
#else
    // Sync fallback: cooperative staging, batched to overlap loads.
    {
        const int tid = wave * 32 + lane;
        const int sub = tid & 3;          // which 16B of the 64B row chunk
        const int nb  = tid >> 2;         // 0..63
        uint4 tmp[4];
#pragma unroll
        for (int o = 0; o < 4; ++o) {
#pragma unroll
            for (int i = 0; i < 4; ++i) {
                int n = nb + 64 * (4 * o + i);
                tmp[i] = *(const uint4*)(wt + (size_t)n * H + k0 + sub * 8);
            }
#pragma unroll
            for (int i = 0; i < 4; ++i) {
                int n = nb + 64 * (4 * o + i);
                *(uint4*)(ldsCur + n * KSTEP + sub * 8) = tmp[i];
            }
        }
    }
#endif
    __syncthreads();   // current tile resident in LDS for all waves

    // Prefetch A for the NEXT K-step; whole compute phase hides the latency.
    if (HN) load_afrag(aNext, y, row0, lane, k0 + KSTEP);

    const int bK = (lane >> 4) * 8;       // B: lane L holds col L%16, K=(L/16)*8 + {0..7} and {16..23}
    const int nBase = (lane & 15) + 128 * wave;

    AB b0, b1;
    b0.u[0] = *(const uint4*)(ldsCur + nBase * KSTEP + bK);
    b0.u[1] = *(const uint4*)(ldsCur + nBase * KSTEP + bK + 16);
#pragma unroll
    for (int j = 0; j < 8; ++j) {
        AB& cur = (j & 1) ? b1 : b0;
        AB& nxt = (j & 1) ? b0 : b1;
        if (j < 7) {
            int n = nBase + 16 * (j + 1);
            nxt.u[0] = *(const uint4*)(ldsCur + n * KSTEP + bK);
            nxt.u[1] = *(const uint4*)(ldsCur + n * KSTEP + bK + 16);
        }
        acc[0][j] = __builtin_amdgcn_wmma_f32_16x16x32_bf16(
            false, aCur[0].v, false, cur.v, (short)0, acc[0][j], false, false);
        acc[1][j] = __builtin_amdgcn_wmma_f32_16x16x32_bf16(
            false, aCur[1].v, false, cur.v, (short)0, acc[1][j], false, false);
    }
    __syncthreads();   // all waves done reading ldsCur; safe to overwrite next step
}

// ---------------------------------------------------------------------------
// GEMM (bf16 WMMA, f32 acc) + fused add-rmsnorm epilogue.
// Block = 256 threads = 8 waves; tile = 32 rows x full N=1024 (fused row-norm).
// Double-buffered B tiles staged by the Tensor Data Mover; A double-buffered
// in registers.
// ---------------------------------------------------------------------------
__global__ __launch_bounds__(256) void gemm_fused_kernel(
    const bf16* __restrict__ wt,    // [N=H][K=H] bf16 (W^T)
    const float* __restrict__ g,    // rmsnorm gamma for this layer's output
    bf16* __restrict__ y,           // in: A rows (bf16); out: next-layer input (in place)
    float* __restrict__ resid,      // in/out residual stream (= d_out)
    float* __restrict__ outF,       // fp32 final output (valid when last)
    int last)
{
    __shared__ __align__(16) bf16 ldsB0[H * KSTEP];   // 64 KB
    __shared__ __align__(16) bf16 ldsB1[H * KSTEP];   // 64 KB
    __shared__ float lds_sq[TROWS];

    const int tid  = threadIdx.x;
    const int lane = tid & 31;
    const int wave = tid >> 5;
    const int row0 = blockIdx.x * TROWS;

    v8f acc[2][8];
#pragma unroll
    for (int gi = 0; gi < 2; ++gi)
#pragma unroll
        for (int j = 0; j < 8; ++j)
            acc[gi][j] = v8f{0.f, 0.f, 0.f, 0.f, 0.f, 0.f, 0.f, 0.f};

#if HAVE_TDM
    if (wave == 0) tdm_load_btile(wt, ldsB0);         // prologue: B tile 0
#endif
    AB afA[2], afB[2];
    load_afrag(afA, y, row0, lane, 0);                // prologue: A chunk 0

    int k0 = 0;
    for (; k0 + 2 * KSTEP < H; k0 += 2 * KSTEP) {
        gemm_step<1>(y, wt, ldsB0, ldsB1, k0,         row0, lane, wave, afA, afB, acc);
        gemm_step<1>(y, wt, ldsB1, ldsB0, k0 + KSTEP, row0, lane, wave, afB, afA, acc);
    }
    gemm_step<1>(y, wt, ldsB0, ldsB1, k0,         row0, lane, wave, afA, afB, acc);
    gemm_step<0>(y, wt, ldsB1, ldsB0, k0 + KSTEP, row0, lane, wave, afB, afA, acc);

    // ---- fused add-rmsnorm epilogue (32 rows) ----
    if (tid < TROWS) lds_sq[tid] = 0.f;
    __syncthreads();

    const int mBase = (lane >> 4) * 8;   // C/D layout: VGPR r -> row mBase + r
#pragma unroll
    for (int gi = 0; gi < 2; ++gi) {
        float sq[8];
#pragma unroll
        for (int r = 0; r < 8; ++r) sq[r] = 0.f;
#pragma unroll
        for (int j = 0; j < 8; ++j) {
            int n = (lane & 15) + 16 * j + 128 * wave;
#pragma unroll
            for (int r = 0; r < 8; ++r) {
                size_t idx = (size_t)(row0 + 16 * gi + mBase + r) * H + n;
                float s = acc[gi][j][r] + resid[idx];
                acc[gi][j][r] = s;
                sq[r] += s * s;
                if (!last) resid[idx] = s;
            }
        }
#pragma unroll
        for (int r = 0; r < 8; ++r) atomicAdd(&lds_sq[16 * gi + mBase + r], sq[r]);
    }
    __syncthreads();

#pragma unroll
    for (int gi = 0; gi < 2; ++gi) {
        float scale[8];
#pragma unroll
        for (int r = 0; r < 8; ++r)
            scale[r] = rsqrtf(lds_sq[16 * gi + mBase + r] * (1.0f / H) + EPS);
#pragma unroll
        for (int j = 0; j < 8; ++j) {
            int n = (lane & 15) + 16 * j + 128 * wave;
            float gv = g[n];
#pragma unroll
            for (int r = 0; r < 8; ++r) {
                size_t idx = (size_t)(row0 + 16 * gi + mBase + r) * H + n;
                float yv = acc[gi][j][r] * scale[r] * gv;
                if (last) outF[idx] = yv;
                else      y[idx] = (bf16)yv;
            }
        }
    }
}

// ---------------------------------------------------------------------------
// Host launch
// ---------------------------------------------------------------------------
extern "C" void kernel_launch(void* const* d_in, const int* in_sizes, int n_in,
                              void* d_out, int out_size, void* d_ws, size_t ws_size,
                              hipStream_t stream) {
    const float* x  = (const float*)d_in[0];
    const float* w0 = (const float*)d_in[1];
    const float* w1 = (const float*)d_in[2];
    const float* w2 = (const float*)d_in[3];
    const float* g0 = (const float*)d_in[4];
    const float* g1 = (const float*)d_in[5];
    const float* g2 = (const float*)d_in[6];
    const float* g3 = (const float*)d_in[7];
    float* out = (float*)d_out;

    const int T = 65536;

    // Workspace layout: y bf16 (T*H*2 = 128MB), then 3x W^T bf16 (2MB each).
    bf16* ybuf = (bf16*)d_ws;
    bf16* wt0  = (bf16*)((char*)d_ws + (size_t)T * H * sizeof(bf16));
    bf16* wt1  = wt0 + (size_t)H * H;
    bf16* wt2  = wt1 + (size_t)H * H;

    wt_kernel<<<(H * H) / 256, 256, 0, stream>>>(w0, wt0);
    wt_kernel<<<(H * H) / 256, 256, 0, stream>>>(w1, wt1);
    wt_kernel<<<(H * H) / 256, 256, 0, stream>>>(w2, wt2);

    // relu + rmsnorm; resid stream lives in d_out
    relu_rms_kernel<<<T, 256, 0, stream>>>(x, g0, out, ybuf);

    gemm_fused_kernel<<<T / TROWS, 256, 0, stream>>>(wt0, g1, ybuf, out, out, 0);
    gemm_fused_kernel<<<T / TROWS, 256, 0, stream>>>(wt1, g2, ybuf, out, out, 0);
    gemm_fused_kernel<<<T / TROWS, 256, 0, stream>>>(wt2, g3, ybuf, out, out, 1);
}
